// Model_46188078301807
// MI455X (gfx1250) — matrix operand order repaired and checked by reading, not compiled, not run on hardware
//
#include <hip/hip_runtime.h>
#include <hip/hip_bf16.h>
#include <math.h>

typedef _Float16 h16;
typedef __attribute__((ext_vector_type(16))) _Float16 v16h;
typedef __attribute__((ext_vector_type(8)))  float    v8f;

#define EPS_BN 1e-5f

// ---------------------------------------------------------------------------
// small helper kernels
// ---------------------------------------------------------------------------
// x: f32 [3][D*H*W] (NCDHW) -> f16 NHWC padded to 4 channels: dst[m*4 + c]
__global__ void k_x_pad(const float* __restrict__ src, h16* __restrict__ dst, size_t DHW) {
  size_t N = DHW * 4;
  size_t stride = (size_t)gridDim.x * blockDim.x;
  for (size_t i = (size_t)blockIdx.x * blockDim.x + threadIdx.x; i < N; i += stride) {
    int c = (int)(i & 3);
    size_t m = i >> 2;
    dst[i] = (h16)(c < 3 ? src[(size_t)c * DHW + m] : 0.0f);
  }
}

__global__ void k_bn_prep(const float* __restrict__ g, const float* __restrict__ b,
                          const float* __restrict__ m, const float* __restrict__ v,
                          float* __restrict__ s, float* __restrict__ t, int C) {
  int i = blockIdx.x * blockDim.x + threadIdx.x;
  if (i < C) { float sc = g[i] * rsqrtf(v[i] + EPS_BN); s[i] = sc; t[i] = b[i] - m[i] * sc; }
}

// weights f32 [Cout][CinReal][ntap] -> f16 [Kpad][Cout], K ordered k = tap*Cinp + ci,
// zero-padded in both the channel-pad region (ci >= CinReal) and K-pad region.
__global__ void k_wconv(const float* __restrict__ src, h16* __restrict__ dst,
                        int Cout, int CinReal, int Cinp, int ntap, int Kpad) {
  int i = blockIdx.x * blockDim.x + threadIdx.x;
  int total = Kpad * Cout;
  if (i < total) {
    int k = i / Cout, co = i - k * Cout;
    int Kdim = Cinp * ntap;
    float val = 0.0f;
    if (k < Kdim) {
      int ci = k % Cinp, tap = k / Cinp;
      if (ci < CinReal) val = src[((size_t)co * CinReal + ci) * ntap + tap];
    }
    dst[i] = (h16)val;
  }
}

// ---------------------------------------------------------------------------
// generic implicit-GEMM 3D conv via v_wmma_f32_16x16x32_f16 (NHWC activations)
//   C[M=spatial, N=Cout] = A[M, K=ntap*Cin] * B[K, N],  k = tap*Cin + ci
//   mode 0: out_f16 = relu(conv + bias)*s + t          (conv1/conv2 path)
//   mode 1: out_f32 = (conv + bias)*s + t              (tri first branch)
//   mode 2: out_f32 += (conv + bias)*s + t             (tri other branches)
//   fastA: Cin % 32 == 0 -> each 16-half A load is one contiguous 32B segment
// ---------------------------------------------------------------------------
__global__ __launch_bounds__(256)
void k_conv_wmma(const h16* __restrict__ in, const h16* __restrict__ wt,
                 const float* __restrict__ bias, const float* __restrict__ bns,
                 const float* __restrict__ bnt,
                 h16* __restrict__ outH, float* __restrict__ outF,
                 int Cin, int Din, int Hin, int Win,
                 int Dout, int Hout, int Wout, int Cout,
                 int kd, int kh, int kw, int sd, int sh, int sw,
                 int pd, int ph, int pw, int Kdim, int Kpad, int mode, int fastA)
{
  __shared__ __align__(16) h16 As[128][40];  // 128 x 32 (+8 pad halves)
  __shared__ __align__(16) h16 Bs[32][72];   // 32 x 64 (+8 pad halves)

  const int tid  = threadIdx.x;
  const int wave = tid >> 5;        // 0..7 -> M subtile
  const int lane = tid & 31;
  const int g    = lane >> 4;       // lane group (ISA 16-bit WMMA layout)
  const int l15  = lane & 15;
  const int M    = Dout * Hout * Wout;
  const int m0   = blockIdx.x * 128;
  const int n0   = blockIdx.y * 64;

  // A-tile load assignment: thread -> (row, 16 consecutive k's)
  const int ar  = tid >> 1;
  const int ac0 = (tid & 1) << 4;
  const int mrow = m0 + ar;
  const bool mvalid = mrow < M;
  int od = 0, oh = 0, ow = 0;
  if (mvalid) { ow = mrow % Wout; int t2 = mrow / Wout; oh = t2 % Hout; od = t2 / Hout; }
  const int iz0 = od * sd - pd, iy0 = oh * sh - ph, ix0 = ow * sw - pw;

  // B-tile load assignment: thread -> (k-row, 8 consecutive cols) = one b128
  const int br  = tid >> 3;
  const int bc0 = (tid & 7) << 3;

  v8f acc[4] = {};   // 4 N-subtiles per wave

  for (int kc = 0; kc < Kpad; kc += 32) {
    // ---- B tile: contiguous 16B per thread ----
    const h16* bp = wt + (size_t)(kc + br) * Cout + (size_t)(n0 + bc0);
    *(uint4*)&Bs[br][bc0] = *(const uint4*)bp;
    if (kc + 32 < Kpad)  // prefetch next chunk's B tile (global_prefetch_b8)
      __builtin_prefetch(wt + (size_t)(kc + 32 + br) * Cout + (size_t)(n0 + bc0), 0, 0);

    // ---- A tile ----
    if (fastA) {
      // 32-wide K-chunk lies inside a single spatial tap (Cin % 32 == 0):
      // 16 consecutive halves = one contiguous 32B segment.
      int kk0 = kc + ac0;
      int tap = kk0 / Cin;
      int ci0 = kk0 - tap * Cin;
      int kx = tap % kw; int tt = tap / kw;
      int ky = tt % kh;  int kz = tt / kh;
      uint4 v0 = {}, v1 = {};
      if (mvalid) {
        int iz = iz0 + kz, iy = iy0 + ky, ix = ix0 + kx;
        if ((unsigned)iz < (unsigned)Din && (unsigned)iy < (unsigned)Hin &&
            (unsigned)ix < (unsigned)Win) {
          const uint4* p = (const uint4*)(in + (((size_t)iz * Hin + iy) * Win + ix) * Cin + ci0);
          v0 = p[0];
          v1 = p[1];
        }
      }
      *(uint4*)&As[ar][ac0]     = v0;
      *(uint4*)&As[ar][ac0 + 8] = v1;
    } else {
      // per-element gather (conv1: Cin padded to 4)
      int kcur = kc + ac0;
      int ci = kcur % Cin; int tap = kcur / Cin;
      int kx = tap % kw; int tt = tap / kw;
      int ky = tt % kh;  int kz = tt / kh;
      #pragma unroll
      for (int j = 0; j < 16; ++j) {
        h16 v = (h16)0.0f;
        if (mvalid && (kcur + j) < Kdim) {
          int iz = iz0 + kz, iy = iy0 + ky, ix = ix0 + kx;
          if ((unsigned)iz < (unsigned)Din && (unsigned)iy < (unsigned)Hin &&
              (unsigned)ix < (unsigned)Win)
            v = in[(((size_t)iz * Hin + iy) * Win + ix) * Cin + ci];
        }
        As[ar][ac0 + j] = v;
        ++ci;
        if (ci == Cin) { ci = 0; ++kx;
          if (kx == kw) { kx = 0; ++ky;
            if (ky == kh) { ky = 0; ++kz; } } }
      }
    }
    __syncthreads();

    // ---- fragments per ISA 7.12.2 ----
    // A 16x32: lanes0-15 row=l15 hold K {0..7,16..23}; lanes16-31 hold {8..15,24..31}
    v16h a;
    #pragma unroll
    for (int h = 0; h < 16; ++h) {
      int kk = h + ((h >= 8) ? 8 : 0) + g * 8;
      a[h] = As[wave * 16 + l15][kk];
    }
    // B 32x16: lanes0-15 col=l15 hold K {0..7,16..23}; lanes16-31 hold {8..15,24..31}
    v16h b0, b1, b2, b3;
    #pragma unroll
    for (int h = 0; h < 16; ++h) {
      int kk = h + ((h >= 8) ? 8 : 0) + g * 8;
      b0[h] = Bs[kk][ 0 + l15];
      b1[h] = Bs[kk][16 + l15];
      b2[h] = Bs[kk][32 + l15];
      b3[h] = Bs[kk][48 + l15];
    }
    acc[0] = __builtin_amdgcn_wmma_f32_16x16x32_f16(false, a, false, b0, (short)0, acc[0], false, false);
    acc[1] = __builtin_amdgcn_wmma_f32_16x16x32_f16(false, a, false, b1, (short)0, acc[1], false, false);
    acc[2] = __builtin_amdgcn_wmma_f32_16x16x32_f16(false, a, false, b2, (short)0, acc[2], false, false);
    acc[3] = __builtin_amdgcn_wmma_f32_16x16x32_f16(false, a, false, b3, (short)0, acc[3], false, false);
    __syncthreads();
  }

  // ---- epilogue: C/D layout VGPR r -> M = r (lanes0-15) / r+8 (lanes16-31) ----
  // NHWC output: lanes write 16 consecutive channels -> coalesced segments.
  #pragma unroll
  for (int ni = 0; ni < 4; ++ni) {
    const int nn = n0 + ni * 16 + l15;
    const float bb = bias[nn], ss = bns[nn], sh2 = bnt[nn];
    #pragma unroll
    for (int r = 0; r < 8; ++r) {
      int mm = m0 + wave * 16 + r + g * 8;
      if (mm < M) {
        float y = acc[ni][r] + bb;
        size_t o = (size_t)mm * Cout + nn;
        if (mode == 0) {
          y = fmaxf(y, 0.0f) * ss + sh2;
          outH[o] = (h16)y;
        } else {
          y = y * ss + sh2;
          if (mode == 2) outF[o] += y; else outF[o] = y;
        }
      }
    }
  }
}

// relu + 2x2x2 maxpool on NHWC, f32 -> f16 (starting max at 0 == fused ReLU)
__global__ void k_relu_maxpool(const float* __restrict__ in, h16* __restrict__ out,
                               int C, int D, int H, int W) {
  int Do = D >> 1, Ho = H >> 1, Wo = W >> 1;
  size_t N = (size_t)C * Do * Ho * Wo;
  size_t stride = (size_t)gridDim.x * blockDim.x;
  for (size_t i = (size_t)blockIdx.x * blockDim.x + threadIdx.x; i < N; i += stride) {
    int c = (int)(i % C); size_t t = i / C;
    int wo = (int)(t % Wo); t /= Wo;
    int ho = (int)(t % Ho); int d0 = (int)(t / Ho);
    const float* p = in + ((((size_t)d0 * 2 * H + (size_t)ho * 2) * W + (size_t)wo * 2) * C + c);
    const size_t sZ = (size_t)H * W * C, sY = (size_t)W * C, sX = C;
    float m = 0.0f;
    #pragma unroll
    for (int dz = 0; dz < 2; ++dz)
      #pragma unroll
      for (int dy = 0; dy < 2; ++dy)
        #pragma unroll
        for (int dx = 0; dx < 2; ++dx)
          m = fmaxf(m, p[dz * sZ + dy * sY + dx * sX]);
    out[i] = (h16)m;
  }
}

// FC over NHWC act [15][14][14][128]; reference flatten is NCDHW -> permute w index.
__global__ void k_fc_sigmoid(const h16* __restrict__ act, const float* __restrict__ w,
                             const float* __restrict__ b, float* __restrict__ out, int N) {
  __shared__ float red[256];
  float s = 0.0f;
  for (int i = threadIdx.x; i < N; i += blockDim.x) {
    int c = i & 127; int t = i >> 7;
    int x = t % 14; t /= 14;
    int y = t % 14; int d = t / 14;
    int widx = ((c * 15 + d) * 14 + y) * 14 + x;
    s += (float)act[i] * w[widx];
  }
  red[threadIdx.x] = s;
  __syncthreads();
  for (int off = blockDim.x >> 1; off > 0; off >>= 1) {
    if ((int)threadIdx.x < off) red[threadIdx.x] += red[threadIdx.x + off];
    __syncthreads();
  }
  if (threadIdx.x == 0) out[0] = 1.0f / (1.0f + expf(-(red[0] + b[0])));
}

// ---------------------------------------------------------------------------
// host side
// ---------------------------------------------------------------------------
// d_in order (setup_inputs insertion order; bn dicts as g,b,m,v):
enum {
  IN_X = 0,
  IN_C1_W, IN_C1_B, IN_BN1_G, IN_BN1_B, IN_BN1_M, IN_BN1_V,
  IN_C2_W, IN_C2_B, IN_BN2_G, IN_BN2_B, IN_BN2_M, IN_BN2_V,
  IN_C3HW_W, IN_C3HW_B, IN_BN3HW_G, IN_BN3HW_B, IN_BN3HW_M, IN_BN3HW_V,
  IN_C3LW_W, IN_C3LW_B, IN_BN3LW_G, IN_BN3LW_B, IN_BN3LW_M, IN_BN3LW_V,
  IN_C3LH_W, IN_C3LH_B, IN_BN3LH_G, IN_BN3LH_B, IN_BN3LH_M, IN_BN3LH_V,
  IN_C4HW_W, IN_C4HW_B, IN_BN4HW_G, IN_BN4HW_B, IN_BN4HW_M, IN_BN4HW_V,
  IN_C4LW_W, IN_C4LW_B, IN_BN4LW_G, IN_BN4LW_B, IN_BN4LW_M, IN_BN4LW_V,
  IN_C4LH_W, IN_C4LH_B, IN_BN4LH_G, IN_BN4LH_B, IN_BN4LH_M, IN_BN4LH_V,
  IN_FC_W, IN_FC_B
};

static inline size_t aln256(size_t x) { return (x + 255) & ~(size_t)255; }

static void launch_conv(hipStream_t s,
                        const h16* in, const h16* wt, const float* bias,
                        const float* bns, const float* bnt,
                        h16* outH, float* outF,
                        int Cin, int Din, int Hin, int Win,
                        int Dout, int Hout, int Wout, int Cout,
                        int kd, int kh, int kw, int sd, int sh, int sw,
                        int pd, int ph, int pw, int Kdim, int Kpad, int mode) {
  int M = Dout * Hout * Wout;
  int fastA = (Cin % 32 == 0) ? 1 : 0;
  dim3 grid((M + 127) / 128, Cout / 64);
  k_conv_wmma<<<grid, 256, 0, s>>>(in, wt, bias, bns, bnt, outH, outF,
                                   Cin, Din, Hin, Win, Dout, Hout, Wout, Cout,
                                   kd, kh, kw, sd, sh, sw, pd, ph, pw,
                                   Kdim, Kpad, mode, fastA);
}

extern "C" void kernel_launch(void* const* d_in, const int* in_sizes, int n_in,
                              void* d_out, int out_size, void* d_ws, size_t ws_size,
                              hipStream_t stream) {
  (void)in_sizes; (void)n_in; (void)out_size; (void)ws_size;
  char* ws = (char*)d_ws;
  size_t off = 0;
  // region0: x_f16 NHWC padded to 4ch (96.3MB) then reused as y3 f32 accumulator (96.3MB)
  const size_t R0 = off;      off += aln256(96337920);
  // region1: act1 f16 (192.7MB) then reused as y4 f32 accumulator (12MB)
  const size_t R1 = off;      off += aln256(192675840);
  const size_t oACT2 = off;   off += aln256(12042240ull * 2);
  const size_t oACT3 = off;   off += aln256(3010560ull * 2);
  const size_t oACT4 = off;   off += aln256(376320ull * 2);
  const size_t oW1   = off;   off += aln256(128ull  * 64  * 2);
  const size_t oW2   = off;   off += aln256(1728ull * 64  * 2);
  const size_t oW3HW = off;   off += aln256(576ull  * 128 * 2);
  const size_t oW3LW = off;   off += aln256(576ull  * 128 * 2);
  const size_t oW3LH = off;   off += aln256(576ull  * 128 * 2);
  const size_t oW4HW = off;   off += aln256(1152ull * 128 * 2);
  const size_t oW4LW = off;   off += aln256(1152ull * 128 * 2);
  const size_t oW4LH = off;   off += aln256(1152ull * 128 * 2);
  const size_t oBNS  = off;   off += aln256(8ull * 128 * 4);
  const size_t oBNT  = off;   off += aln256(8ull * 128 * 4);

  h16*   xh   = (h16*)(ws + R0);
  float* y3   = (float*)(ws + R0);
  h16*   act1 = (h16*)(ws + R1);
  float* y4   = (float*)(ws + R1);
  h16*   act2 = (h16*)(ws + oACT2);
  h16*   act3 = (h16*)(ws + oACT3);
  h16*   act4 = (h16*)(ws + oACT4);
  float* bnS  = (float*)(ws + oBNS);
  float* bnT  = (float*)(ws + oBNT);

  const float* x = (const float*)d_in[IN_X];

  // input f32 NCDHW -> f16 NHWC padded to 4 channels
  k_x_pad<<<2048, 256, 0, stream>>>(x, xh, 12042240ull /* 240*224*224 */);

  // BN scale/shift: sets 0..7 = bn1,bn2,bn3_hw,bn3_lw,bn3_lh,bn4_hw,bn4_lw,bn4_lh
  const int bnBase[8] = {IN_BN1_G, IN_BN2_G, IN_BN3HW_G, IN_BN3LW_G,
                         IN_BN3LH_G, IN_BN4HW_G, IN_BN4LW_G, IN_BN4LH_G};
  const int bnC[8]    = {64, 64, 128, 128, 128, 128, 128, 128};
  for (int i = 0; i < 8; ++i) {
    k_bn_prep<<<1, 128, 0, stream>>>((const float*)d_in[bnBase[i] + 0],
                                     (const float*)d_in[bnBase[i] + 1],
                                     (const float*)d_in[bnBase[i] + 2],
                                     (const float*)d_in[bnBase[i] + 3],
                                     bnS + i * 128, bnT + i * 128, bnC[i]);
  }

  // weight transpose/pad to [Kpad][Cout] f16, K = tap*Cinp + ci
  k_wconv<<<(128 * 64 + 255) / 256, 256, 0, stream>>>((const float*)d_in[IN_C1_W],   (h16*)(ws + oW1),   64, 3, 4, 27, 128);
  k_wconv<<<(1728 * 64 + 255) / 256, 256, 0, stream>>>((const float*)d_in[IN_C2_W],  (h16*)(ws + oW2),   64, 64, 64, 27, 1728);
  k_wconv<<<(576 * 128 + 255) / 256, 256, 0, stream>>>((const float*)d_in[IN_C3HW_W], (h16*)(ws + oW3HW), 128, 64, 64, 9, 576);
  k_wconv<<<(576 * 128 + 255) / 256, 256, 0, stream>>>((const float*)d_in[IN_C3LW_W], (h16*)(ws + oW3LW), 128, 64, 64, 9, 576);
  k_wconv<<<(576 * 128 + 255) / 256, 256, 0, stream>>>((const float*)d_in[IN_C3LH_W], (h16*)(ws + oW3LH), 128, 64, 64, 9, 576);
  k_wconv<<<(1152 * 128 + 255) / 256, 256, 0, stream>>>((const float*)d_in[IN_C4HW_W], (h16*)(ws + oW4HW), 128, 128, 128, 9, 1152);
  k_wconv<<<(1152 * 128 + 255) / 256, 256, 0, stream>>>((const float*)d_in[IN_C4LW_W], (h16*)(ws + oW4LW), 128, 128, 128, 9, 1152);
  k_wconv<<<(1152 * 128 + 255) / 256, 256, 0, stream>>>((const float*)d_in[IN_C4LH_W], (h16*)(ws + oW4LH), 128, 128, 128, 9, 1152);

  // conv1: 4(pad) -> 64, 3x3x3 s2 p1, bn(relu(.)) -> act1 f16 NHWC
  launch_conv(stream, xh, (h16*)(ws + oW1), (const float*)d_in[IN_C1_B],
              bnS + 0 * 128, bnT + 0 * 128, act1, nullptr,
              4, 240, 224, 224, 120, 112, 112, 64,
              3, 3, 3, 2, 2, 2, 1, 1, 1, 108, 128, 0);

  // conv2: 64 -> 64, 3x3x3 s2 p1 -> act2 f16 NHWC
  launch_conv(stream, act1, (h16*)(ws + oW2), (const float*)d_in[IN_C2_B],
              bnS + 1 * 128, bnT + 1 * 128, act2, nullptr,
              64, 120, 112, 112, 60, 56, 56, 64,
              3, 3, 3, 2, 2, 2, 1, 1, 1, 1728, 1728, 0);

  // tri block 3: 64 -> 128 on (60,56,56); hw(1,3,3) + lw(3,1,3) + lh(3,3,1) -> y3 f32
  launch_conv(stream, act2, (h16*)(ws + oW3HW), (const float*)d_in[IN_C3HW_B],
              bnS + 2 * 128, bnT + 2 * 128, nullptr, y3,
              64, 60, 56, 56, 60, 56, 56, 128,
              1, 3, 3, 1, 1, 1, 0, 1, 1, 576, 576, 1);
  launch_conv(stream, act2, (h16*)(ws + oW3LW), (const float*)d_in[IN_C3LW_B],
              bnS + 3 * 128, bnT + 3 * 128, nullptr, y3,
              64, 60, 56, 56, 60, 56, 56, 128,
              3, 1, 3, 1, 1, 1, 1, 0, 1, 576, 576, 2);
  launch_conv(stream, act2, (h16*)(ws + oW3LH), (const float*)d_in[IN_C3LH_B],
              bnS + 4 * 128, bnT + 4 * 128, nullptr, y3,
              64, 60, 56, 56, 60, 56, 56, 128,
              3, 3, 1, 1, 1, 1, 1, 1, 0, 576, 576, 2);
  k_relu_maxpool<<<4096, 256, 0, stream>>>(y3, act3, 128, 60, 56, 56);

  // tri block 4: 128 -> 128 on (30,28,28) -> y4 f32
  launch_conv(stream, act3, (h16*)(ws + oW4HW), (const float*)d_in[IN_C4HW_B],
              bnS + 5 * 128, bnT + 5 * 128, nullptr, y4,
              128, 30, 28, 28, 30, 28, 28, 128,
              1, 3, 3, 1, 1, 1, 0, 1, 1, 1152, 1152, 1);
  launch_conv(stream, act3, (h16*)(ws + oW4LW), (const float*)d_in[IN_C4LW_B],
              bnS + 6 * 128, bnT + 6 * 128, nullptr, y4,
              128, 30, 28, 28, 30, 28, 28, 128,
              3, 1, 3, 1, 1, 1, 1, 0, 1, 1152, 1152, 2);
  launch_conv(stream, act3, (h16*)(ws + oW4LH), (const float*)d_in[IN_C4LH_B],
              bnS + 7 * 128, bnT + 7 * 128, nullptr, y4,
              128, 30, 28, 28, 30, 28, 28, 128,
              3, 3, 1, 1, 1, 1, 1, 1, 0, 1152, 1152, 2);
  k_relu_maxpool<<<1024, 256, 0, stream>>>(y4, act4, 128, 30, 28, 28);

  // FC (376320 -> 1) + sigmoid (NHWC -> NCDHW weight permute)
  k_fc_sigmoid<<<1, 256, 0, stream>>>(act4, (const float*)d_in[IN_FC_W],
                                      (const float*)d_in[IN_FC_B], (float*)d_out, 376320);
}
